// ColBERTPairwise_77695958384873
// MI455X (gfx1250) — matrix operand order repaired and checked by reading, not compiled, not run on hardware
//
#include <hip/hip_runtime.h>
#include <hip/hip_bf16.h>

typedef __attribute__((ext_vector_type(16))) _Float16 v16h;
typedef __attribute__((ext_vector_type(8)))  _Float16 v8h;
typedef __attribute__((ext_vector_type(4)))  _Float16 v4h;
typedef __attribute__((ext_vector_type(8)))  float    v8f;

// exp(ALPHA*x) = exp2(A2*x);  log(S)/ALPHA = log2(S)*L2
#define A2 17.312340490667561f   // 12 * log2(e)
#define L2 0.05776226504666211f  // ln(2) / 12
#define NEG_INF (-__builtin_inff())

// Native base-2 transcendentals: v_exp_f32 / v_log_f32, no wrapper code.
#if __has_builtin(__builtin_amdgcn_exp2f)
#define EXP2F(x) __builtin_amdgcn_exp2f(x)
#else
#define EXP2F(x) exp2f(x)
#endif
#if __has_builtin(__builtin_amdgcn_logf)
#define LOG2F(x) __builtin_amdgcn_logf(x)
#else
#define LOG2F(x) log2f(x)
#endif

// ---------------------------------------------------------------------------
// Pass 1: L2-normalize rows of [nrows, 128] f32 -> f16.  One wave per row.
// ---------------------------------------------------------------------------
__global__ __launch_bounds__(256) void l2norm_to_f16(
    const float* __restrict__ x, _Float16* __restrict__ y, int nrows)
{
    const int row  = blockIdx.x * 8 + (threadIdx.x >> 5);
    const int lane = threadIdx.x & 31;
    if (row >= nrows) return;

    const float4 v = ((const float4*)(x + (size_t)row * 128))[lane];
    float ss = v.x * v.x + v.y * v.y + v.z * v.z + v.w * v.w;
#pragma unroll
    for (int off = 16; off > 0; off >>= 1) ss += __shfl_xor(ss, off);
    const float inv = 1.0f / fmaxf(sqrtf(ss), 1e-12f);

    v4h h;
    h[0] = (_Float16)(v.x * inv);
    h[1] = (_Float16)(v.y * inv);
    h[2] = (_Float16)(v.z * inv);
    h[3] = (_Float16)(v.w * inv);
    ((v4h*)(y + (size_t)row * 128))[lane] = h;
}

// ---------------------------------------------------------------------------
// Pass 2: per (b,o): S = qn[b] (32x128) @ kn[o]^T (128x256) via WMMA f16,
// fused masked logsumexp-pool over Lk, sum over Lq, scale.  Block = 8 waves.
// Wave w: m-half = w&1 (16 rows), column group g = w>>1 (64 cols, 4 N-tiles).
// Cosine similarities are bounded by ~1, so we use a FIXED softmax max of
// 1.0: lse = 1 + log2(sum exp2(A2*(s-1)))*L2.  No max reduction, no NaN
// guards; masked cols get bias -inf -> exp2 -> 0; all-masked row -> S=0 ->
// log2(0) = -inf, matching the reference exactly.
// ---------------------------------------------------------------------------
__global__ __launch_bounds__(256) void colbert_pairwise_wmma(
    const _Float16* __restrict__ qh,     // [64*32, 128]
    const _Float16* __restrict__ kh,     // [128*256, 128]
    const unsigned char* __restrict__ qmask,  // [64, 32]   1 = padding
    const unsigned char* __restrict__ kmask,  // [128, 256] 1 = padding
    const float* __restrict__ logit_scale,
    float* __restrict__ out)             // [64, 128]
{
    const int o    = blockIdx.x;         // 0..127
    const int b    = blockIdx.y;         // 0..63
    const int tid  = threadIdx.x;
    const int wave = tid >> 5;
    const int lane = tid & 31;
    const int mh   = wave & 1;           // which 16-row half of Lq
    const int g    = wave >> 1;          // 64-column group of Lk
    const int l16  = lane & 15;
    const int lhi  = lane >> 4;

    // Block-uniform bases (SGPRs); all per-lane offsets are 32-bit.
    const _Float16* qbase = qh + (b * 32 + mh * 16) * 128;
    const _Float16* kbase = kh + o * 256 * 128;
    const unsigned char* kmbase = kmask + o * 256 + g * 64;

    // A fragments for this wave's 16 rows, all 4 K-steps (16x32 f16 each).
    // Lane layout: M = l16; K = 32*ks + lhi*8 + {e (e<8) , e+8 (e>=8)}.
    v16h A[4];
#pragma unroll
    for (int ks = 0; ks < 4; ++ks) {
        const int off = (l16 << 7) + ks * 32 + lhi * 8;
        v8h lo = *(const v8h*)(qbase + off);
        v8h hi = *(const v8h*)(qbase + off + 16);
#pragma unroll
        for (int e = 0; e < 8; ++e) { A[ks][e] = lo[e]; A[ks][e + 8] = hi[e]; }
    }

    // Compute and keep all 4 C tiles (32 VGPRs) + per-lane exp bias.
    v8f   C[4];
    float w[4];   // A2 * bias: masked -> -inf, else -A2 (the fixed max 1.0)
#pragma unroll
    for (int nt = 0; nt < 4; ++nt) {
        w[nt] = (kmbase[nt * 16 + l16] != 0) ? NEG_INF : -A2;
        v8f c = {};
#pragma unroll
        for (int ks = 0; ks < 4; ++ks) {
            // B layout: N = l16 (row of kn), K = 32*ks + lhi*8 + {e (e<8) , e+8 (e>=8)}.
            const int off = ((g * 64 + nt * 16 + l16) << 7) + ks * 32 + lhi * 8;
            v8h blo = *(const v8h*)(kbase + off);
            v8h bhi = *(const v8h*)(kbase + off + 16);
            v16h Bf;
#pragma unroll
            for (int e = 0; e < 8; ++e) { Bf[e] = blo[e]; Bf[e + 8] = bhi[e]; }
            c = __builtin_amdgcn_wmma_f32_16x16x32_f16(
                    false, A[ks], false, Bf, (short)0, c, false, false);
        }
        C[nt] = c;
    }

    // One fma + one exp2 per S element (32 per lane), masked via bias.
    float S[8];
#pragma unroll
    for (int r = 0; r < 8; ++r) {
        S[r] = EXP2F(fmaf(A2, C[0][r], w[0]))
             + EXP2F(fmaf(A2, C[1][r], w[1]))
             + EXP2F(fmaf(A2, C[2][r], w[2]))
             + EXP2F(fmaf(A2, C[3][r], w[3]));
    }
    // Sum-only reduction across the 16 lanes holding one row's columns.
#pragma unroll
    for (int off = 1; off < 16; off <<= 1)
#pragma unroll
        for (int r = 0; r < 8; ++r)
            S[r] += __shfl_xor(S[r], off);

    // Per-row partial sums for the 4 column groups.
    __shared__ float part[32][4];
    if (l16 == 0) {
        const int rowbase = mh * 16 + lhi * 8;
#pragma unroll
        for (int r = 0; r < 8; ++r)
            part[rowbase + r][g] = S[r];
    }
    __syncthreads();

    if (wave == 0) {
        const int row = lane;                      // 0..31
        const float S4 = part[row][0] + part[row][1]
                       + part[row][2] + part[row][3];
        // lse(alpha*S)/alpha = 1 + log2(S4)*ln2/alpha ; S4==0 -> -inf.
        float lse = 1.0f + LOG2F(S4) * L2;
        if (qmask[b * 32 + row] != 0) lse = NEG_INF;

        // Sum over the 32 query positions (-inf propagates like the ref).
#pragma unroll
        for (int off = 1; off < 32; off <<= 1) lse += __shfl_xor(lse, off);

        if (lane == 0) {
            // where(isfinite, s, 0): NaN/-inf -> 0
            const float total =
                (lse > NEG_INF && lse < __builtin_inff()) ? lse : 0.0f;
            const float temp  = fminf(__expf(logit_scale[0]), 100.0f);
            const float norm  = 1.0f / (sqrtf(32.0f * 256.0f) + 1e-6f);
            out[b * 128 + o] = total * norm * temp;
        }
    }
}

// ---------------------------------------------------------------------------
extern "C" void kernel_launch(void* const* d_in, const int* in_sizes, int n_in,
                              void* d_out, int out_size, void* d_ws, size_t ws_size,
                              hipStream_t stream) {
    const float*         q  = (const float*)d_in[0];          // [64,32,128]
    const float*         k  = (const float*)d_in[1];          // [128,256,128]
    const unsigned char* qm = (const unsigned char*)d_in[2];  // [64,32] bool
    const unsigned char* km = (const unsigned char*)d_in[3];  // [128,256] bool
    const float*         ls = (const float*)d_in[4];          // scalar
    float*               out = (float*)d_out;                 // [64,128]

    const int QROWS = 64 * 32;    // 2048
    const int KROWS = 128 * 256;  // 32768

    _Float16* qh = (_Float16*)d_ws;
    _Float16* kh = (_Float16*)((char*)d_ws + (size_t)QROWS * 128 * sizeof(_Float16));

    // Normalize + downconvert (bandwidth-trivial: ~17 MB total).
    l2norm_to_f16<<<QROWS / 8, 256, 0, stream>>>(q, qh, QROWS);
    l2norm_to_f16<<<KROWS / 8, 256, 0, stream>>>(k, kh, KROWS);

    // One block per (o, b) pair; 8 wave32 per block; 16 WMMA per wave.
    dim3 grid(128, 64);
    colbert_pairwise_wmma<<<grid, 256, 0, stream>>>(qh, kh, qm, km, ls, out);
}
